// lstm_32057635897779
// MI455X (gfx1250) — matrix operand order repaired and checked by reading, not compiled, not run on hardware
//
#include <hip/hip_runtime.h>

// ---------------------------------------------------------------------------
// Problem dims
// ---------------------------------------------------------------------------
typedef __bf16 bf16_t;
typedef __attribute__((ext_vector_type(16))) __bf16 v16bf;
typedef __attribute__((ext_vector_type(8)))  float  v8f;

union Frag16 { v16bf v; uint4 u[2]; };   // 32 bytes = 16 bf16 per lane

__device__ __forceinline__ v8f wmma_bf16(v16bf a, v16bf b, v8f c) {
  // D = A(16x32 bf16) * B(32x16 bf16) + C(16x16 f32)
  return __builtin_amdgcn_wmma_f32_16x16x32_bf16(false, a, false, b, (short)0, c,
                                                 false, false);
}
__device__ __forceinline__ float sigf(float x) { return 1.0f / (1.0f + __expf(-x)); }
__device__ __forceinline__ v8f vzero8() {
  v8f z;
#pragma unroll
  for (int i = 0; i < 8; ++i) z[i] = 0.f;
  return z;
}

constexpr int VOC = 32000, EMB = 128, HID = 128, SEQ = 50, BAT = 256;
constexpr int NP = 64;           // padded LSTM "batch" rows (50 -> 64)
constexpr int G4 = 512;          // 4*H gate width
constexpr int KC = 256;          // concat K = E + H
constexpr int HS = 6400;         // H*S (K of logits GEMM)
constexpr int AS = KC + 8;       // bf16 LDS row stride for A buffers (264)
constexpr int GS = G4 + 8;       // f32  LDS row stride for gates (520)

constexpr size_t LOGN   = (size_t)BAT * VOC;     // 8,192,000 logits
constexpr size_t HN_OFF = LOGN;                  // h_n [2,50,128]
constexpr size_t CN_OFF = LOGN + 2 * SEQ * HID;  // c_n [2,50,128]

// dynamic LDS for the LSTM kernel: A0,A1 (bf16) + gates,c0,c1 (f32)
constexpr int LSTM_LDS = (2 * NP * AS) * 2 + (NP * GS + 2 * NP * HID) * 4; // 266,240 B

// workspace layout (bytes)
constexpr size_t WS_XBF  = 0;                                  // [256][64][128] bf16
constexpr size_t WS_WT   = WS_XBF + (size_t)BAT * NP * EMB * 2; // [2][512][256] bf16
constexpr size_t WS_BIAS = WS_WT + (size_t)2 * KC * G4 * 2;     // [2][512] f32
constexpr size_t WS_OUTT = WS_BIAS + (size_t)2 * G4 * 4;        // [256][6400] bf16

// ---------------------------------------------------------------------------
// Prep: W^T concat + bf16 convert.  wt[l][n][k] = (k<128 ? Wih : Whh)[n][k%128]
// ---------------------------------------------------------------------------
__global__ void prep_wt_kernel(const float* __restrict__ Wih0, const float* __restrict__ Whh0,
                               const float* __restrict__ Wih1, const float* __restrict__ Whh1,
                               bf16_t* __restrict__ wt) {
  int l = blockIdx.x >> 8, k = blockIdx.x & 255;
  const float* W = (l == 0) ? ((k < 128) ? Wih0 : Whh0) : ((k < 128) ? Wih1 : Whh1);
  int kk = k & 127;
  bf16_t* dst = wt + (size_t)l * KC * G4 + k;
  for (int n = threadIdx.x; n < G4; n += blockDim.x)
    dst[(size_t)n * KC] = (bf16_t)W[(size_t)n * 128 + kk];
}

__global__ void prep_bias_kernel(const float* __restrict__ bih0, const float* __restrict__ bhh0,
                                 const float* __restrict__ bih1, const float* __restrict__ bhh1,
                                 float* __restrict__ bias) {
  int i = threadIdx.x;  // 512
  bias[i]      = bih0[i] + bhh0[i];
  bias[G4 + i] = bih1[i] + bhh1[i];
}

// ---------------------------------------------------------------------------
// Embedding gather -> padded bf16 x buffer [256][64][128]
// ---------------------------------------------------------------------------
__global__ void embed_kernel(const int* __restrict__ words, const float* __restrict__ table,
                             bf16_t* __restrict__ xbf) {
  int t = blockIdx.x;           // 0..255 (time = original batch index)
  int h = threadIdx.x;          // 0..127
  for (int m = 0; m < NP; ++m) {
    float v = 0.f;
    if (m < SEQ) {
      int w = words[t * SEQ + m];
      v = table[(size_t)w * EMB + h];
    }
    xbf[((size_t)t * NP + m) * EMB + h] = (bf16_t)v;
  }
}

// ---------------------------------------------------------------------------
// Persistent single-workgroup 2-layer LSTM. 512 threads = 16 wave32.
// Per step, per layer: gates[64x512] = [x|h](64x256 bf16) @ WT(256x512 bf16)
//   each wave owns 2 N-tiles (32 gate cols) across all 4 M-tiles.
// ---------------------------------------------------------------------------
__global__ __launch_bounds__(512, 1)
void lstm_kernel(const bf16_t* __restrict__ xbf, const bf16_t* __restrict__ wt,
                 const float* __restrict__ bias, bf16_t* __restrict__ outT,
                 float* __restrict__ dout) {
  extern __shared__ char smem[];
  bf16_t* A0    = (bf16_t*)smem;                       // [64][264] = [x_t | h0]
  bf16_t* A1    = A0 + NP * AS;                        // [64][264] = [h0 | h1]
  float*  gates = (float*)(smem + (size_t)2 * NP * AS * 2);  // [64][520]
  float*  c0    = gates + NP * GS;                     // [64][128]
  float*  c1    = c0 + NP * HID;

  const int tid  = threadIdx.x;
  const int lane = tid & 31;
  const int wave = tid >> 5;     // 0..15
  const int hi16 = (lane >> 4) & 1;

  // zero-init all LDS (h regions of A0/A1 and c must start at 0)
  for (int i = tid; i < LSTM_LDS / 4; i += 512) ((unsigned int*)smem)[i] = 0u;
  __syncthreads();

  for (int t = 0; t < BAT; ++t) {
    // stage x_t into A0[:, 0:128] (1024 uint4, 2 per thread)
    const uint4* src = (const uint4*)(xbf + (size_t)t * NP * EMB);
#pragma unroll
    for (int i = 0; i < 2; ++i) {
      int v4  = tid + i * 512;       // 16 uint4 per 128-bf16 row
      int row = v4 >> 4;
      int c4  = v4 & 15;
      ((uint4*)(A0 + row * AS))[c4] = src[v4];
    }
    if (t + 1 < BAT) __builtin_prefetch(xbf + (size_t)(t + 1) * NP * EMB, 0, 0);
    __syncthreads();

#pragma unroll 1
    for (int layer = 0; layer < 2; ++layer) {
      const bf16_t* Abuf = (layer == 0) ? A0 : A1;
      const bf16_t* WTl  = wt + (size_t)layer * KC * G4;
      const float*  bl   = bias + layer * G4;
      float*        cbuf = (layer == 0) ? c0 : c1;

      v8f acc[4][2];
#pragma unroll
      for (int mt = 0; mt < 4; ++mt) { acc[mt][0] = vzero8(); acc[mt][1] = vzero8(); }

      const int nt0 = wave * 2;
#pragma unroll
      for (int kt = 0; kt < 8; ++kt) {
        // B fragments: lane holds n = nt*16 + (lane&15), K-chunks {kb..kb+7, kb+16..kb+23}
        const int kb = kt * 32 + hi16 * 8;
        const bf16_t* bp0 = WTl + (size_t)(nt0 * 16 + (lane & 15)) * KC + kb;
        const bf16_t* bp1 = bp0 + (size_t)16 * KC;
        Frag16 b0, b1;
        b0.u[0] = *(const uint4*)(bp0);
        b0.u[1] = *(const uint4*)(bp0 + 16);
        b1.u[0] = *(const uint4*)(bp1);
        b1.u[1] = *(const uint4*)(bp1 + 16);
#pragma unroll
        for (int mt = 0; mt < 4; ++mt) {
          int m = mt * 16 + (lane & 15);
          Frag16 a;  // K-chunks {kb..kb+7, kb+16..kb+23}
          a.u[0] = *(const uint4*)(Abuf + m * AS + kb);
          a.u[1] = *(const uint4*)(Abuf + m * AS + kb + 16);
          acc[mt][0] = wmma_bf16(a.v, b0.v, acc[mt][0]);
          acc[mt][1] = wmma_bf16(a.v, b1.v, acc[mt][1]);
        }
      }

      // gates (+bias) -> LDS.  D: m = mt*16 + v + 8*hi16, n = nt*16 + (lane&15)
#pragma unroll
      for (int mt = 0; mt < 4; ++mt) {
#pragma unroll
        for (int j = 0; j < 2; ++j) {
          int n  = (nt0 + j) * 16 + (lane & 15);
          float bv = bl[n];
          int mb = mt * 16 + hi16 * 8;
#pragma unroll
          for (int v = 0; v < 8; ++v)
            gates[(mb + v) * GS + n] = acc[mt][j][v] + bv;
        }
      }
      __syncthreads();

      // elementwise LSTM cell; 64x128 elems over 512 threads
      for (int idx = tid; idx < NP * HID; idx += 512) {
        int m = idx >> 7, hc = idx & 127;
        float gi = gates[m * GS + hc];
        float gf = gates[m * GS + 128 + hc];
        float gg = gates[m * GS + 256 + hc];
        float go = gates[m * GS + 384 + hc];
        float cn = sigf(gf) * cbuf[idx] + sigf(gi) * tanhf(gg);
        float hn = sigf(go) * tanhf(cn);
        cbuf[idx] = cn;
        bf16_t hb = (bf16_t)hn;
        if (layer == 0) {
          A0[m * AS + 128 + hc] = hb;   // h0 for next step
          A1[m * AS + hc]       = hb;   // input to layer 1
        } else {
          A1[m * AS + 128 + hc] = hb;   // h1 for next step
          if (m < SEQ) outT[(size_t)t * HS + m * HID + hc] = hb;  // out [t][k]
        }
        if (t == BAT - 1 && m < SEQ) {
          dout[HN_OFF + (size_t)layer * SEQ * HID + m * HID + hc] = hn;
          dout[CN_OFF + (size_t)layer * SEQ * HID + m * HID + hc] = cn;
        }
      }
      __syncthreads();
    }
  }
}

// ---------------------------------------------------------------------------
// logits^T = W_lin(32000x6400 f32, cvt->bf16 in-regs) @ out^T(6400x256 bf16)
// Block: 256 thr (8 waves), tile 128 vocab x 256 t.  Wave w: mt=0..7, nt={2w,2w+1}.
// W_lin streamed once from HBM (819 MB -> ~35us floor), staged via LDS.
// ---------------------------------------------------------------------------
__global__ __launch_bounds__(256, 1)
void logits_kernel(const float* __restrict__ Wlin, const float* __restrict__ blin,
                   const bf16_t* __restrict__ outT, float* __restrict__ dout) {
  __shared__ bf16_t Asm[128 * 40];  // [128 vocab rows][32 k] bf16, stride 40

  const int tid  = threadIdx.x;
  const int lane = tid & 31;
  const int wave = tid >> 5;     // 0..7
  const int hi16 = (lane >> 4) & 1;
  const int mv0  = blockIdx.x * 128;

  v8f acc[8][2];
#pragma unroll
  for (int mt = 0; mt < 8; ++mt) { acc[mt][0] = vzero8(); acc[mt][1] = vzero8(); }

  const int row = tid >> 1;            // cooperative A-load: 2 threads/row
  const int kq  = (tid & 1) * 16;

#pragma unroll 1
  for (int kt = 0; kt < HS / 32; ++kt) {
    // ---- stage [128][32] f32 -> bf16 LDS (coalesced 64B per thread) ----
    const float4* src = (const float4*)(Wlin + (size_t)(mv0 + row) * HS + kt * 32 + kq);
    float4 f0 = src[0], f1 = src[1], f2 = src[2], f3 = src[3];
    if (kt + 1 < HS / 32)
      __builtin_prefetch(Wlin + (size_t)(mv0 + row) * HS + (kt + 1) * 32 + kq, 0, 0);
    bf16_t* dr = Asm + row * 40 + kq;
    dr[0]=(bf16_t)f0.x; dr[1]=(bf16_t)f0.y; dr[2]=(bf16_t)f0.z; dr[3]=(bf16_t)f0.w;
    dr[4]=(bf16_t)f1.x; dr[5]=(bf16_t)f1.y; dr[6]=(bf16_t)f1.z; dr[7]=(bf16_t)f1.w;
    dr[8]=(bf16_t)f2.x; dr[9]=(bf16_t)f2.y; dr[10]=(bf16_t)f2.z; dr[11]=(bf16_t)f2.w;
    dr[12]=(bf16_t)f3.x; dr[13]=(bf16_t)f3.y; dr[14]=(bf16_t)f3.z; dr[15]=(bf16_t)f3.w;
    __syncthreads();

    // ---- B fragments from out [t][k]: lane t = nt*16+(lane&15), K-chunks {kb..kb+7, kb+16..kb+23} ----
    const bf16_t* bp0 = outT + (size_t)(wave * 32 + (lane & 15)) * HS + kt * 32 + hi16 * 8;
    const bf16_t* bp1 = bp0 + (size_t)16 * HS;
    Frag16 b0, b1;
    b0.u[0] = *(const uint4*)(bp0);
    b0.u[1] = *(const uint4*)(bp0 + 16);
    b1.u[0] = *(const uint4*)(bp1);
    b1.u[1] = *(const uint4*)(bp1 + 16);

    const int kb = hi16 * 8;
#pragma unroll
    for (int mt = 0; mt < 8; ++mt) {
      int m = mt * 16 + (lane & 15);
      Frag16 a;
      a.u[0] = *(const uint4*)(Asm + m * 40 + kb);
      a.u[1] = *(const uint4*)(Asm + m * 40 + kb + 16);
      acc[mt][0] = wmma_bf16(a.v, b0.v, acc[mt][0]);
      acc[mt][1] = wmma_bf16(a.v, b1.v, acc[mt][1]);
    }
    __syncthreads();
  }

  // ---- epilogue: D(m=vocab, n=t) -> logits[t][vocab], contiguous 32B/lane ----
#pragma unroll
  for (int mt = 0; mt < 8; ++mt) {
#pragma unroll
    for (int j = 0; j < 2; ++j) {
      int tIdx = (wave * 2 + j) * 16 + (lane & 15);
      int vb   = mv0 + mt * 16 + hi16 * 8;
      float4 lo, hi;
      lo.x = acc[mt][j][0] + blin[vb + 0];
      lo.y = acc[mt][j][1] + blin[vb + 1];
      lo.z = acc[mt][j][2] + blin[vb + 2];
      lo.w = acc[mt][j][3] + blin[vb + 3];
      hi.x = acc[mt][j][4] + blin[vb + 4];
      hi.y = acc[mt][j][5] + blin[vb + 5];
      hi.z = acc[mt][j][6] + blin[vb + 6];
      hi.w = acc[mt][j][7] + blin[vb + 7];
      float* out = dout + (size_t)tIdx * VOC + vb;
      *(float4*)(out)     = lo;
      *(float4*)(out + 4) = hi;
    }
  }
}

// ---------------------------------------------------------------------------
extern "C" void kernel_launch(void* const* d_in, const int* in_sizes, int n_in,
                              void* d_out, int out_size, void* d_ws, size_t ws_size,
                              hipStream_t stream) {
  (void)in_sizes; (void)n_in; (void)out_size; (void)ws_size;
  const int*   words = (const int*)  d_in[0];
  const float* table = (const float*)d_in[1];
  const float* Wih0  = (const float*)d_in[2];
  const float* Whh0  = (const float*)d_in[3];
  const float* bih0  = (const float*)d_in[4];
  const float* bhh0  = (const float*)d_in[5];
  const float* Wih1  = (const float*)d_in[6];
  const float* Whh1  = (const float*)d_in[7];
  const float* bih1  = (const float*)d_in[8];
  const float* bhh1  = (const float*)d_in[9];
  const float* Wlin  = (const float*)d_in[10];
  const float* blin  = (const float*)d_in[11];
  float* dout = (float*)d_out;

  char* ws = (char*)d_ws;
  bf16_t* xbf  = (bf16_t*)(ws + WS_XBF);
  bf16_t* wt   = (bf16_t*)(ws + WS_WT);
  float*  bias = (float*) (ws + WS_BIAS);
  bf16_t* outT = (bf16_t*)(ws + WS_OUTT);

  (void)hipFuncSetAttribute((const void*)lstm_kernel,
                            hipFuncAttributeMaxDynamicSharedMemorySize, LSTM_LDS);

  prep_wt_kernel  <<<512, 128, 0, stream>>>(Wih0, Whh0, Wih1, Whh1, wt);
  prep_bias_kernel<<<1,   512, 0, stream>>>(bih0, bhh0, bih1, bhh1, bias);
  embed_kernel    <<<256, 128, 0, stream>>>(words, table, xbf);
  lstm_kernel     <<<1,   512, LSTM_LDS, stream>>>(xbf, wt, bias, outT, dout);
  logits_kernel   <<<250, 256, 0, stream>>>(Wlin, blin, outT, dout);
}
